// NeuralSDE_21852793602428
// MI455X (gfx1250) — matrix operand order repaired and checked by reading, not compiled, not run on hardware
//
#include <hip/hip_runtime.h>
#include <hip/hip_bf16.h>

// ---------------------------------------------------------------------------
// Neural SDE (ReversibleHeun) for MI455X / gfx1250, wave32 + WMMA bf16.
// One persistent kernel: each 256-thread (8-wave) WG owns a 64-row batch tile
// and runs the entire T=64 scan. All GEMMs via v_wmma_f32_16x16x32_bf16.
// Weights pre-converted to bf16 in d_ws (gW1b = 1MB -> L2 resident).
// LDS activation tiles padded to odd multiples of 16B to kill bank conflicts.
// ---------------------------------------------------------------------------

typedef __attribute__((ext_vector_type(16))) __bf16 v16bf;
typedef __attribute__((ext_vector_type(8)))  __bf16 v8bf;
typedef __attribute__((ext_vector_type(8)))  float  v8f;

#define B_TOT  8192
#define HDIM   64
#define NDIM   32
#define WDIM   256
#define TSTEPS 64
#define DDIM   16
#define IDIM   64
#define MT     64      // batch rows per workgroup
#define NWAVE  8
#define BLOCK  256

#define XLD    72      // padded LDS stride for [64,64] bf16 tiles (9 x 16B, odd)
#define HLD    264     // padded LDS stride for [64,256] bf16 tile (33 x 16B, odd)
#define XIDX(i) ((((i) >> 6) * XLD) + ((i) & 63))

__device__ __forceinline__ __bf16 f2bf(float x) { return (__bf16)x; }  // native cvt, RNE

__device__ __forceinline__ float fast_tanh(float x) {
  float e = __expf(2.f * x);           // +inf -> 1, 0 -> -1 handled correctly
  return 1.f - 2.f / (e + 1.f);
}

// One 16x16 f32 tile of  A(LDS bf16, row-major, LDA) x B^T(global bf16 [N,K] row-major, LDW)
// A frag (16x32): lane l (half=l>>4, r=l&15) holds row m0+r, k = kb + half*8 + {0..7, 16..23}
// B frag (32x16): lane l holds col n0+r, k = kb + half*8 + {0..7, 16..23}
template<int LDA, int LDW, int K>
__device__ __forceinline__ v8f tile_gemm(const __bf16* sA,
                                         const __bf16* __restrict__ W,
                                         int m0, int n0, v8f acc) {
  const int lane = threadIdx.x & 31;
  const int half = lane >> 4, r = lane & 15;
  const __bf16* arow = sA + (m0 + r) * LDA + half * 8;
  const __bf16* wrow = W + (size_t)(n0 + r) * (size_t)LDW + half * 8;
#pragma unroll
  for (int kb = 0; kb < K; kb += 32) {
    v8bf a0 = *(const v8bf*)(arow + kb);
    v8bf a1 = *(const v8bf*)(arow + kb + 16);
    v8bf b0 = *(const v8bf*)(wrow + kb);
    v8bf b1 = *(const v8bf*)(wrow + kb + 16);
    v16bf a, b;
#pragma unroll
    for (int i = 0; i < 8; ++i) { a[i] = a0[i]; a[i + 8] = a1[i];
                                  b[i] = b0[i]; b[i + 8] = b1[i]; }
    acc = __builtin_amdgcn_wmma_f32_16x16x32_bf16(false, a, false, b,
                                                  (short)0, acc, false, false);
  }
  return acc;
}

// x[64,64] @ W^T  (W bf16 [256,64]) + bias(+t*wcol0), relu(0)/lipswish(1) -> sH bf16 [64,HLD]
__device__ void layer_hidden(const __bf16* sX, const __bf16* __restrict__ Wb,
                             const float* __restrict__ bias,
                             const float* __restrict__ wcol0, float tcur,
                             __bf16* sH, int act) {
  const int wave = threadIdx.x >> 5, lane = threadIdx.x & 31;
  const int half = lane >> 4, r = lane & 15;
  for (int t = wave; t < (MT / 16) * (WDIM / 16); t += NWAVE) {
    const int m0 = (t & 3) << 4, n0 = (t >> 2) << 4;
    v8f acc = {};
    acc = tile_gemm<XLD, HDIM, HDIM>(sX, Wb, m0, n0, acc);
    float b = bias[n0 + r];
    if (wcol0) b += tcur * wcol0[n0 + r];
#pragma unroll
    for (int v = 0; v < 8; ++v) {
      float x = acc[v] + b;
      float y = (act == 0) ? fmaxf(x, 0.f)
                           : 0.909f * x / (1.f + __expf(-x));
      sH[(m0 + v + 8 * half) * HLD + n0 + r] = f2bf(y);
    }
  }
}

// sH[64,HLD] @ W^T (W bf16 [64,256]) + bias, optional tanh -> sOut f32 [64,64] flat
// optionally also mirror into padded bf16 buffers (for init y0).
__device__ void layer_out64(const __bf16* sH, const __bf16* __restrict__ Wb,
                            const float* __restrict__ bias, float* sOut,
                            int do_tanh, __bf16* sXout, __bf16* sYout) {
  const int wave = threadIdx.x >> 5, lane = threadIdx.x & 31;
  const int half = lane >> 4, r = lane & 15;
  for (int t = wave; t < (MT / 16) * (HDIM / 16); t += NWAVE) {
    const int m0 = (t & 3) << 4, n0 = (t >> 2) << 4;
    v8f acc = {};
    acc = tile_gemm<HLD, WDIM, WDIM>(sH, Wb, m0, n0, acc);
    const float b = bias[n0 + r];
#pragma unroll
    for (int v = 0; v < 8; ++v) {
      float x = acc[v] + b;
      if (do_tanh) x = fast_tanh(x);
      const int m = m0 + v + 8 * half, h = n0 + r;
      sOut[m * HDIM + h] = x;
      if (sXout) sXout[m * XLD + h] = f2bf(x);
      if (sYout) sYout[m * XLD + h] = f2bf(x);
    }
  }
}

// diffusion output: sH[64,HLD] @ gW1^T (bf16 [2048,256]) + gb1, tanh,
// contracted on the fly (shfl_xor tree over the 16-lane halves, then 1 atomic):
//   sEpC += fg1 . dw0 ,  sEpN += fg1 . dw1
__device__ void layer_diff_out(const __bf16* sH, const __bf16* __restrict__ gW1b,
                               const float* __restrict__ gb1,
                               const float* sDw0, const float* sDw1,
                               float* sEpC, float* sEpN, int have_next) {
  const int wave = threadIdx.x >> 5, lane = threadIdx.x & 31;
  const int half = lane >> 4, r = lane & 15;
  for (int t = wave; t < (MT / 16) * ((HDIM * NDIM) / 16); t += NWAVE) { // 512 tiles
    const int m0 = (t & 3) << 4, n0 = (t >> 2) << 4;
    v8f acc = {};
    acc = tile_gemm<HLD, WDIM, WDIM>(sH, gW1b, m0, n0, acc);
    const float b = gb1[n0 + r];
    const int hcol = n0 >> 5;            // h = o / 32  (constant per tile)
    const int nmod = (n0 & 16) + r;      // n = o % 32
#pragma unroll
    for (int v = 0; v < 8; ++v) {
      const int m = m0 + v + 8 * half;
      const float g = fast_tanh(acc[v] + b);
      float c0 = g * sDw0[m * NDIM + nmod];
#pragma unroll
      for (int s = 8; s >= 1; s >>= 1) c0 += __shfl_xor(c0, s, 32);
      if (r == 0) atomicAdd(&sEpC[m * HDIM + hcol], c0);
      if (have_next) {
        float c1 = g * sDw1[m * NDIM + nmod];
#pragma unroll
        for (int s = 8; s >= 1; s >>= 1) c1 += __shfl_xor(c1, s, 32);
        if (r == 0) atomicAdd(&sEpN[m * HDIM + hcol], c1);
      }
    }
  }
}

// y[64,XLD] @ rW^T (bf16 [16,64]) + rb -> global f32 [64,16]
__device__ void layer_readout(const __bf16* sYbf, const __bf16* __restrict__ rWb,
                              const float* __restrict__ rb,
                              float* __restrict__ outp) {
  const int wave = threadIdx.x >> 5, lane = threadIdx.x & 31;
  const int half = lane >> 4, r = lane & 15;
  for (int t = wave; t < MT / 16; t += NWAVE) {   // 4 tiles -> waves 0..3
    const int m0 = t << 4;
    v8f acc = {};
    acc = tile_gemm<XLD, HDIM, HDIM>(sYbf, rWb, m0, 0, acc);
    const float b = rb[r];
#pragma unroll
    for (int v = 0; v < 8; ++v)
      outp[(size_t)(m0 + v + 8 * half) * DDIM + r] = acc[v] + b;
  }
}

// -------------------- weight prep (f32 -> bf16 in d_ws) --------------------
__global__ void cvt_bf16(const float* __restrict__ src, __bf16* __restrict__ dst, int n) {
  int i = blockIdx.x * 256 + threadIdx.x;
  if (i < n) dst[i] = f2bf(src[i]);
}
// split [256,65] -> bf16 [256,64] (cols 1..64) + f32 col0[256]
__global__ void split_w0(const float* __restrict__ src, __bf16* __restrict__ dst,
                         float* __restrict__ col0) {
  int i = blockIdx.x * 256 + threadIdx.x;
  if (i < 256 * 64) {
    int o = i >> 6, k = i & 63;
    dst[i] = f2bf(src[o * 65 + k + 1]);
  }
  if (i < 256) col0[i] = src[i * 65];
}

// ------------------------------- main kernel -------------------------------
__global__ __launch_bounds__(BLOCK) void sde_kernel(
    const float* __restrict__ ib0, const float* __restrict__ ib1,
    const float* __restrict__ db0, const float* __restrict__ db1,
    const float* __restrict__ gb0, const float* __restrict__ gb1,
    const float* __restrict__ rb,
    const float* __restrict__ init_noise, const float* __restrict__ bm,
    const float* __restrict__ ts,
    const __bf16* __restrict__ iW0b, const __bf16* __restrict__ iW1b,
    const __bf16* __restrict__ dW0b, const __bf16* __restrict__ dW1b,
    const __bf16* __restrict__ gW0b, const __bf16* __restrict__ gW1b,
    const __bf16* __restrict__ rWb,
    const float* __restrict__ dW0c0, const float* __restrict__ gW0c0,
    float* __restrict__ out) {
  extern __shared__ char smem_raw[];
  float*  sFd  = (float*)smem_raw;          // [64,64] drift output (flat)
  float*  sEpC = sFd  + MT * HDIM;          // [64,64] eprod (current step)
  float*  sEpN = sEpC + MT * HDIM;          // [64,64] eprod (next step)
  float*  sDw0 = sEpN + MT * HDIM;          // [64,32] bm[i]
  float*  sDw1 = sDw0 + MT * NDIM;          // [64,32] bm[i+1]
  __bf16* sH   = (__bf16*)(sDw1 + MT * NDIM);   // [64,HLD] hidden acts
  __bf16* sXbf = sH   + MT * HLD;           // [64,XLD] GEMM input (yhat / noise)
  __bf16* sYbf = sXbf + MT * XLD;           // [64,XLD] y for readout

  const int tid   = threadIdx.x;
  const int bbase = blockIdx.x * MT;

  float yreg[16], yhreg[16], p0reg[16];

  // ---- stage: noise tile, zero next-eprod, dw0 = bm[0]
#pragma unroll
  for (int j = 0; j < 16; ++j) {
    const int idx = tid * 16 + j;
    sXbf[XIDX(idx)] = f2bf(init_noise[(size_t)bbase * IDIM + idx]);
    sEpN[idx] = 0.f;
  }
#pragma unroll
  for (int j = 0; j < 8; ++j) {
    const int idx = tid * 8 + j;
    sDw0[idx] = bm[(size_t)bbase * NDIM + idx];
  }
  __syncthreads();

  // ---- init MLP: relu hidden, linear out -> y0
  layer_hidden(sXbf, iW0b, ib0, nullptr, 0.f, sH, 0);
  __syncthreads();
  layer_out64(sH, iW1b, ib1, sEpC /*temp y0*/, 0, sXbf, sYbf);
  __syncthreads();
  const float t0 = ts[0];
#pragma unroll
  for (int j = 0; j < 16; ++j) {
    const int idx = tid * 16 + j;
    yreg[j]  = sEpC[idx];
    yhreg[j] = yreg[j];
  }
  __syncthreads();

  // ---- fd0 = drift(t0, y0)
  layer_hidden(sXbf, dW0b, db0, dW0c0, t0, sH, 1);
  __syncthreads();
  layer_out64(sH, dW1b, db1, sFd, 1, nullptr, nullptr);
  __syncthreads();
  // ---- eprod0 = diffusion(t0, y0) . bm[0]
  layer_hidden(sXbf, gW0b, gb0, gW0c0, t0, sH, 1);
  __syncthreads();
  layer_diff_out(sH, gW1b, gb1, sDw0, sDw0, sEpN, sEpN, 0);
  __syncthreads();
  // ---- out[0] = y0 @ rW^T + rb
  layer_readout(sYbf, rWb, rb, out + (size_t)bbase * DDIM);
  __syncthreads();

  // ---- ReversibleHeun scan
  for (int i = 0; i < TSTEPS; ++i) {
    const float tB = ts[i + 1];
    const float dt = tB - ts[i];
    const int have_next = (i + 1 < TSTEPS);

    // elementwise: prod0, yhat1; consume & reset eprod buffers
#pragma unroll
    for (int j = 0; j < 16; ++j) {
      const int idx = tid * 16 + j;
      const float ep = sEpN[idx];
      sEpN[idx] = 0.f;
      sEpC[idx] = 0.f;
      const float pr0 = sFd[idx] * dt + ep;
      const float yh1 = 2.f * yreg[j] - yhreg[j] + pr0;
      p0reg[j] = pr0;
      yhreg[j] = yh1;
      sXbf[XIDX(idx)] = f2bf(yh1);
    }
#pragma unroll
    for (int j = 0; j < 8; ++j) {
      const int idx = tid * 8 + j;
      sDw0[idx] = bm[((size_t)i * B_TOT + bbase) * NDIM + idx];
      if (have_next)
        sDw1[idx] = bm[((size_t)(i + 1) * B_TOT + bbase) * NDIM + idx];
    }
    __syncthreads();

    // fd1 = drift(t1, yhat1)
    layer_hidden(sXbf, dW0b, db0, dW0c0, tB, sH, 1);
    __syncthreads();
    layer_out64(sH, dW1b, db1, sFd, 1, nullptr, nullptr);
    __syncthreads();
    // fg1 = diffusion(t1, yhat1); contract with bm[i] and bm[i+1]
    layer_hidden(sXbf, gW0b, gb0, gW0c0, tB, sH, 1);
    __syncthreads();
    layer_diff_out(sH, gW1b, gb1, sDw0, sDw1, sEpC, sEpN, have_next);
    __syncthreads();

    // y1 = y + 0.5*(prod0 + prod1)
#pragma unroll
    for (int j = 0; j < 16; ++j) {
      const int idx = tid * 16 + j;
      const float pr1 = sFd[idx] * dt + sEpC[idx];
      const float y1  = yreg[j] + 0.5f * (p0reg[j] + pr1);
      yreg[j] = y1;
      sYbf[XIDX(idx)] = f2bf(y1);
    }
    __syncthreads();
    layer_readout(sYbf, rWb, rb,
                  out + ((size_t)(i + 1) * B_TOT + bbase) * DDIM);
    __syncthreads();
  }
}

// ------------------------------ host launcher ------------------------------
extern "C" void kernel_launch(void* const* d_in, const int* in_sizes, int n_in,
                              void* d_out, int out_size, void* d_ws, size_t ws_size,
                              hipStream_t stream) {
  (void)in_sizes; (void)n_in; (void)out_size; (void)ws_size;
  const float* iW0 = (const float*)d_in[0];
  const float* ib0 = (const float*)d_in[1];
  const float* iW1 = (const float*)d_in[2];
  const float* ib1 = (const float*)d_in[3];
  const float* dW0 = (const float*)d_in[4];
  const float* db0 = (const float*)d_in[5];
  const float* dW1 = (const float*)d_in[6];
  const float* db1 = (const float*)d_in[7];
  const float* gW0 = (const float*)d_in[8];
  const float* gb0 = (const float*)d_in[9];
  const float* gW1 = (const float*)d_in[10];
  const float* gb1 = (const float*)d_in[11];
  const float* rW  = (const float*)d_in[12];
  const float* rb  = (const float*)d_in[13];
  const float* init_noise = (const float*)d_in[14];
  const float* bm  = (const float*)d_in[15];
  const float* ts  = (const float*)d_in[16];

  char* ws = (char*)d_ws;
  __bf16* gW1b = (__bf16*)ws; ws += (size_t)2048 * 256 * 2;   // 1 MB, L2 resident
  __bf16* iW0b = (__bf16*)ws; ws += 256 * 64 * 2;
  __bf16* iW1b = (__bf16*)ws; ws += 64 * 256 * 2;
  __bf16* dW0b = (__bf16*)ws; ws += 256 * 64 * 2;
  __bf16* dW1b = (__bf16*)ws; ws += 64 * 256 * 2;
  __bf16* gW0b = (__bf16*)ws; ws += 256 * 64 * 2;
  __bf16* rWb  = (__bf16*)ws; ws += 16 * 64 * 2;
  float*  dW0c0 = (float*)ws; ws += 256 * 4;
  float*  gW0c0 = (float*)ws; ws += 256 * 4;

  cvt_bf16<<<(2048 * 256 + 255) / 256, 256, 0, stream>>>(gW1, gW1b, 2048 * 256);
  cvt_bf16<<<64, 256, 0, stream>>>(iW0, iW0b, 256 * 64);
  cvt_bf16<<<64, 256, 0, stream>>>(iW1, iW1b, 64 * 256);
  cvt_bf16<<<64, 256, 0, stream>>>(dW1, dW1b, 64 * 256);
  cvt_bf16<<<4, 256, 0, stream>>>(rW, rWb, 16 * 64);
  split_w0<<<64, 256, 0, stream>>>(dW0, dW0b, dW0c0);
  split_w0<<<64, 256, 0, stream>>>(gW0, gW0b, gW0c0);

  const size_t smem = (size_t)(3 * MT * HDIM + 2 * MT * NDIM) * sizeof(float) +
                      (size_t)(MT * HLD + 2 * MT * XLD) * sizeof(__bf16); // 117760 B
  (void)hipFuncSetAttribute((const void*)sde_kernel,
                            hipFuncAttributeMaxDynamicSharedMemorySize, (int)smem);
  sde_kernel<<<B_TOT / MT, BLOCK, smem, stream>>>(
      ib0, ib1, db0, db1, gb0, gb1, rb, init_noise, bm, ts,
      iW0b, iW1b, dW0b, dW1b, gW0b, gW1b, rWb, dW0c0, gW0c0, (float*)d_out);
}
